// RNN_7748121002357
// MI455X (gfx1250) — matrix operand order repaired and checked by reading, not compiled, not run on hardware
//
#include <hip/hip_runtime.h>
#include <math.h>

// Problem dims (from reference)
#define BB 64
#define TT 512
#define II 256
#define HH 2048
#define OO 256

typedef __bf16 v16bf __attribute__((ext_vector_type(16)));
typedef __bf16 v8bf  __attribute__((ext_vector_type(8)));
typedef float  v8f   __attribute__((ext_vector_type(8)));

__device__ __forceinline__ unsigned short f2bf(float f) {
    unsigned u = __float_as_uint(f);
    u += 0x7FFFu + ((u >> 16) & 1u);   // round-to-nearest-even
    return (unsigned short)(u >> 16);
}
__device__ __forceinline__ float bf2f(unsigned short h) {
    return __uint_as_float(((unsigned)h) << 16);
}

// ---------------- prep kernels (run once, not perf critical) ----------------

__global__ __launch_bounds__(256) void k_cvt_bf16(const float* __restrict__ in,
                                                  unsigned short* __restrict__ out,
                                                  int n) {
    int i = blockIdx.x * blockDim.x + threadIdx.x;
    if (i < n) out[i] = f2bf(in[i]);
}

// in: [K,N] f32 row-major  ->  out: [N,K] bf16 row-major (K-contiguous per column)
__global__ __launch_bounds__(256) void k_transpose_bf16(const float* __restrict__ in,
                                                        unsigned short* __restrict__ out,
                                                        int K, int N) {
    int i = blockIdx.x * blockDim.x + threadIdx.x;
    if (i < K * N) {
        int k = i / N, n = i % N;
        out[(size_t)n * K + k] = f2bf(in[i]);
    }
}

// ---------------- generic WMMA bf16 GEMM ----------------
// C[M,N] = A[M,K](bf16,row-major) * Bt[N,K](bf16,row-major == B^T), + epilogue.
// mode 0: out bf16 [M,N], += bias[n]                                (h0 encoder)
// mode 1: out bf16, += bias[n], row m=(b*T+t) stored at (t*B+b)     (x_proj)
// mode 2: out bf16, tanh(acc + bf16 addsrc[m,n])                    (RNN step)
// mode 3: out f32,  += bias[n], row m=(t*B+b) stored at (b*T+t)*N+n (decoder)
__global__ __launch_bounds__(256) void k_gemm_wmma(
    const unsigned short* __restrict__ A,
    const unsigned short* __restrict__ Bt,
    const float* __restrict__ bias,
    const unsigned short* __restrict__ addsrc,
    void* __restrict__ out,
    int M, int K, int N, int mode, int Bdim, int Tdim)
{
    const int lane   = threadIdx.x & 31;
    const int wave   = blockIdx.x * (blockDim.x >> 5) + (threadIdx.x >> 5);
    const int npairs = N >> 5;            // 32 columns per wave
    const int total  = (M >> 4) * npairs;
    if (wave >= total) return;            // wave-uniform: EXEC stays all-ones

    const int mt = wave / npairs;
    const int np = wave % npairs;
    const int hi = lane >> 4;             // half-wave selector
    const int lr = lane & 15;

    const unsigned short* Arow = A  + (size_t)(mt * 16 + lr) * K;
    const unsigned short* B0   = Bt + (size_t)(np * 32 + lr) * K;
    const unsigned short* B1   = Bt + (size_t)(np * 32 + 16 + lr) * K;
    const int aoff = hi ? 16 : 0;         // A: K 0..15 / 16..31 per half-wave
    const int boff = hi ? 16 : 0;         // B: K 0..15 / 16..31 per half-wave

    v8f acc0 = {}; v8f acc1 = {};

    for (int k0 = 0; k0 < K; k0 += 32) {
        v8bf alo = *reinterpret_cast<const v8bf*>(Arow + k0 + aoff);
        v8bf ahi = *reinterpret_cast<const v8bf*>(Arow + k0 + aoff + 8);
        v16bf a  = __builtin_shufflevector(alo, ahi,
                       0,1,2,3,4,5,6,7,8,9,10,11,12,13,14,15);
        v16bf b0 = *reinterpret_cast<const v16bf*>(B0 + k0 + boff);
        v16bf b1 = *reinterpret_cast<const v16bf*>(B1 + k0 + boff);
        acc0 = __builtin_amdgcn_wmma_f32_16x16x32_bf16(false, a, false, b0,
                                                       (short)0, acc0, false, false);
        acc1 = __builtin_amdgcn_wmma_f32_16x16x32_bf16(false, a, false, b1,
                                                       (short)0, acc1, false, false);
    }

    const int n0 = np * 32 + lr;
    const int n1 = n0 + 16;
    #pragma unroll
    for (int r = 0; r < 8; ++r) {
        const int mrow = mt * 16 + r + 8 * hi;
        float v0 = acc0[r], v1 = acc1[r];
        if (mode == 0) {
            unsigned short* o = (unsigned short*)out;
            o[(size_t)mrow * N + n0] = f2bf(v0 + bias[n0]);
            o[(size_t)mrow * N + n1] = f2bf(v1 + bias[n1]);
        } else if (mode == 1) {
            // m = b*T + t  ->  store at row t*B + b
            int b = mrow / Tdim, t = mrow % Tdim;
            size_t row = (size_t)t * Bdim + b;
            unsigned short* o = (unsigned short*)out;
            o[row * N + n0] = f2bf(v0 + bias[n0]);
            o[row * N + n1] = f2bf(v1 + bias[n1]);
        } else if (mode == 2) {
            unsigned short* o = (unsigned short*)out;
            size_t base = (size_t)mrow * N;
            o[base + n0] = f2bf(tanhf(v0 + bf2f(addsrc[base + n0])));
            o[base + n1] = f2bf(tanhf(v1 + bf2f(addsrc[base + n1])));
        } else {
            // m = t*B + b -> out[b*T*O + t*O + n]
            int t = mrow / Bdim, b = mrow % Bdim;
            float* o = (float*)out;
            size_t base = ((size_t)b * Tdim + t) * N;
            o[base + n0] = v0 + bias[n0];
            o[base + n1] = v1 + bias[n1];
        }
    }
}

// ---------------- host-side orchestration ----------------

static inline int gemm_blocks(int M, int N) {
    int waves = (M / 16) * (N / 32);
    return (waves + 7) / 8;               // 8 waves (256 threads) per block
}

extern "C" void kernel_launch(void* const* d_in, const int* in_sizes, int n_in,
                              void* d_out, int out_size, void* d_ws, size_t ws_size,
                              hipStream_t stream) {
    const float* input = (const float*)d_in[0];   // [B,T,I]
    const float* start = (const float*)d_in[1];   // [B,I]
    const float* W_enc = (const float*)d_in[2];   // [I,H]
    const float* b_enc = (const float*)d_in[3];   // [H]
    const float* W_in  = (const float*)d_in[4];   // [I,H]
    const float* W_rec = (const float*)d_in[5];   // [H,H]
    const float* b_rnn = (const float*)d_in[6];   // [H]
    const float* W_dec = (const float*)d_in[7];   // [H,O]
    const float* b_dec = (const float*)d_in[8];   // [O]

    // carve scratch (bf16 buffers stored as u16 bit patterns)
    char* ws = (char*)d_ws;
    size_t off = 0;
    auto carve = [&](size_t bytes) {
        char* p = ws + off;
        off += (bytes + 255) & ~(size_t)255;
        return p;
    };
    unsigned short* WrecT = (unsigned short*)carve((size_t)HH * HH * 2); // [H(n),H(k)]
    unsigned short* WinT  = (unsigned short*)carve((size_t)HH * II * 2); // [H,I]
    unsigned short* WencT = (unsigned short*)carve((size_t)HH * II * 2); // [H,I]
    unsigned short* WdecT = (unsigned short*)carve((size_t)OO * HH * 2); // [O,H]
    unsigned short* in_bf = (unsigned short*)carve((size_t)BB * TT * II * 2);
    unsigned short* st_bf = (unsigned short*)carve((size_t)BB * II * 2);
    unsigned short* h0    = (unsigned short*)carve((size_t)BB * HH * 2);
    unsigned short* xproj = (unsigned short*)carve((size_t)TT * BB * HH * 2); // [T,B,H]
    unsigned short* hist  = (unsigned short*)carve((size_t)TT * BB * HH * 2); // [T,B,H]

    // 1) weight transposes + input conversions (bf16)
    {
        int n = II * HH;
        k_transpose_bf16<<<(n + 255) / 256, 256, 0, stream>>>(W_enc, WencT, II, HH);
        k_transpose_bf16<<<(n + 255) / 256, 256, 0, stream>>>(W_in,  WinT,  II, HH);
    }
    {
        int n = HH * HH;
        k_transpose_bf16<<<(n + 255) / 256, 256, 0, stream>>>(W_rec, WrecT, HH, HH);
    }
    {
        int n = HH * OO;
        k_transpose_bf16<<<(n + 255) / 256, 256, 0, stream>>>(W_dec, WdecT, HH, OO);
    }
    {
        int n = BB * TT * II;
        k_cvt_bf16<<<(n + 255) / 256, 256, 0, stream>>>(input, in_bf, n);
        int m = BB * II;
        k_cvt_bf16<<<(m + 255) / 256, 256, 0, stream>>>(start, st_bf, m);
    }

    // 2) h0 = start @ W_enc + b_enc          (M=64, K=256, N=2048)
    k_gemm_wmma<<<gemm_blocks(BB, HH), 256, 0, stream>>>(
        st_bf, WencT, b_enc, nullptr, h0, BB, II, HH, 0, BB, TT);

    // 3) x_proj[t,b,h] = input @ W_in + b_rnn (M=B*T, K=256, N=2048)
    k_gemm_wmma<<<gemm_blocks(BB * TT, HH), 256, 0, stream>>>(
        in_bf, WinT, b_rnn, nullptr, xproj, BB * TT, II, HH, 1, BB, TT);

    // 4) recurrence: h_t = tanh(x_proj_t + h_{t-1} @ W_rec)  (512 sequential steps)
    //    W_rec^T (8.4 MB bf16) stays resident in the 192 MB L2 across steps.
    for (int t = 0; t < TT; ++t) {
        const unsigned short* hprev =
            (t == 0) ? h0 : (hist + (size_t)(t - 1) * BB * HH);
        k_gemm_wmma<<<gemm_blocks(BB, HH), 256, 0, stream>>>(
            hprev, WrecT, nullptr,
            xproj + (size_t)t * BB * HH,
            hist  + (size_t)t * BB * HH,
            BB, HH, HH, 2, BB, TT);
    }

    // 5) out[b,t,:] = h_t @ W_dec + b_dec    (M=T*B, K=2048, N=256, f32 out)
    k_gemm_wmma<<<gemm_blocks(TT * BB, OO), 256, 0, stream>>>(
        hist, WdecT, b_dec, nullptr, d_out, TT * BB, HH, OO, 3, BB, TT);
}
